// Encoder_8701603742155
// MI455X (gfx1250) — matrix operand order repaired and checked by reading, not compiled, not run on hardware
//
#include <hip/hip_runtime.h>
#include <hip/hip_bf16.h>
#include <math.h>

// ---------------- types for CDNA5 WMMA ----------------
typedef __bf16 bf16_t;
typedef __attribute__((ext_vector_type(8)))  __bf16 v8bf;
typedef __attribute__((ext_vector_type(16))) __bf16 v16bf;
typedef __attribute__((ext_vector_type(8)))  float  v8f;

#define WMMA_BF16(a, b, c) \
  __builtin_amdgcn_wmma_f32_16x16x32_bf16(false, (a), false, (b), (short)0, (c), false, false)

static constexpr int Sdim = 32, Hdim = 64, Wdim = 64, Cdim = 192;
static constexpr int HEADS = 6, WSZ = 4, NTOK = 64, HID = 768;
static constexpr int NWIN = (Sdim / WSZ) * (Hdim / WSZ) * (Wdim / WSZ); // 2048
static constexpr int MTOT = Sdim * Hdim * Wdim;                        // 131072

__device__ __forceinline__ bf16_t f2bf(float f) { return (bf16_t)f; }

// A-fragment (16x32 bf16) from row-major A, per ISA layout:
// lane 0-15: row=lane, K = [0..8) u [16..24); lane 16-31: row=lane-16, K=[8..16) u [24..32)
__device__ __forceinline__ v16bf load_afrag(const bf16_t* A, int lda, int lane) {
  int row = lane & 15, kg = lane >> 4;
  const bf16_t* p = A + (long)row * lda + kg * 8;
  v8bf c0 = *(const v8bf*)(p);
  v8bf c1 = *(const v8bf*)(p + 16);
  v16bf f;
#pragma unroll
  for (int i = 0; i < 8; ++i) { f[i] = c0[i]; f[i + 8] = c1[i]; }
  return f;
}

// B-fragment (32x16 bf16) from BT stored as N x K row-major:
// lane 0-15: col=lane, K = [0..8) u [16..24); lane 16-31: col=lane-16, K=[8..16) u [24..32)
__device__ __forceinline__ v16bf load_bfrag(const bf16_t* BT, int ldb, int lane) {
  int col = lane & 15, kg = lane >> 4;
  const bf16_t* p = BT + (long)col * ldb + kg * 8;
  v8bf c0 = *(const v8bf*)(p);
  v8bf c1 = *(const v8bf*)(p + 16);
  v16bf f;
#pragma unroll
  for (int i = 0; i < 8; ++i) { f[i] = c0[i]; f[i + 8] = c1[i]; }
  return f;
}

// A-fragment from an f32 row-major matrix (LDS softmax probs), converted to bf16
__device__ __forceinline__ v16bf load_afrag_f32(const float* A, int lda, int lane, int kbase) {
  int row = lane & 15, kg = lane >> 4;
  const float* p = A + row * lda + kbase + kg * 8;
  v16bf f;
#pragma unroll
  for (int i = 0; i < 8; ++i) f[i] = f2bf(p[i]);
  const float* q = p + 16;
#pragma unroll
  for (int i = 0; i < 8; ++i) f[i + 8] = f2bf(q[i]);
  return f;
}

// ---------------- weight prep: f32 [K,N] -> bf16 [N,K] ----------------
__global__ void k_transpose(const float* __restrict__ w, bf16_t* __restrict__ wt, int K, int N) {
  int idx = blockIdx.x * 256 + threadIdx.x;
  if (idx >= K * N) return;
  int k = idx / N, n = idx % N;
  wt[(long)n * K + k] = f2bf(w[idx]);
}

// ---------------- window partition + roll, f32 -> bf16 ----------------
__global__ void k_gather(const float* __restrict__ x, bf16_t* __restrict__ xw, int shift) {
  int idx = blockIdx.x * 256 + threadIdx.x;           // NWIN*64*192 = 25,165,824
  int wi = idx / (NTOK * Cdim);
  int r  = idx % (NTOK * Cdim);
  int tk = r / Cdim, c = r % Cdim;
  int wis = wi >> 8, wih = (wi >> 4) & 15, wiw = wi & 15;
  int ts = tk >> 4, th = (tk >> 2) & 3, tw = tk & 3;
  int s  = (wis * 4 + ts + shift) & (Sdim - 1);
  int hh = (wih * 4 + th + shift) & (Hdim - 1);
  int ww = (wiw * 4 + tw + shift) & (Wdim - 1);
  int src = (s * Hdim + hh) * Wdim + ww;
  xw[idx] = f2bf(x[(long)src * Cdim + c]);
}

// ---------------- tiled GEMM: block tile 128x64, wave tile 32x32 ----------------
// C = A(bf16 MxK row-major) * BT(bf16 NxK row-major)^T + bias, K compile-time.
template <int GELU, int K>
__global__ __launch_bounds__(256) void k_gemm_tile(const bf16_t* __restrict__ A,
                                                   const bf16_t* __restrict__ BT,
                                                   const float* __restrict__ bias,
                                                   bf16_t* __restrict__ Out, int N) {
  int w = threadIdx.x >> 5, lane = threadIdx.x & 31;
  int wm = w & 3, wn = w >> 2;                     // 4 M-waves x 2 N-waves
  int mrow = blockIdx.x * 128 + wm * 32;
  int ncol = blockIdx.y * 64 + wn * 32;
  v8f acc[2][2] = {};
  const bf16_t* A0 = A + (long)mrow * K;
  const bf16_t* A1 = A0 + (long)16 * K;
  const bf16_t* B0 = BT + (long)ncol * K;
  const bf16_t* B1 = B0 + (long)16 * K;
#pragma unroll 3
  for (int kt = 0; kt < K; kt += 32) {
    v16bf a0 = load_afrag(A0 + kt, K, lane);
    v16bf a1 = load_afrag(A1 + kt, K, lane);
    v16bf b0 = load_bfrag(B0 + kt, K, lane);
    v16bf b1 = load_bfrag(B1 + kt, K, lane);
    acc[0][0] = WMMA_BF16(a0, b0, acc[0][0]);
    acc[0][1] = WMMA_BF16(a0, b1, acc[0][1]);
    acc[1][0] = WMMA_BF16(a1, b0, acc[1][0]);
    acc[1][1] = WMMA_BF16(a1, b1, acc[1][1]);
  }
  int ln = lane & 15, lh = lane >> 4;
#pragma unroll
  for (int mi = 0; mi < 2; ++mi)
#pragma unroll
    for (int nf = 0; nf < 2; ++nf) {
      int col = ncol + nf * 16 + ln;
      float bv = bias[col];
#pragma unroll
      for (int j = 0; j < 8; ++j) {
        int row = mrow + mi * 16 + j + lh * 8;
        float v = acc[mi][nf][j] + bv;
        if (GELU) v = 0.5f * v * (1.f + erff(v * 0.70710678118f));   // exact GELU
        Out[(long)row * N + col] = f2bf(v);
      }
    }
}

// ---------------- full-row GEMM (N=192) + fused res-post-norm LayerNorm ----------------
// SCATTER=1: window-reverse + roll scatter (proj).  SCATTER=0: identity rows (fc2).
template <int SCATTER, int K>
__global__ __launch_bounds__(256) void k_gemm_row_ln(const bf16_t* __restrict__ A,
                                                     const bf16_t* __restrict__ BT,
                                                     const float* __restrict__ bias,
                                                     const float* __restrict__ shortcut,
                                                     const float* __restrict__ gamma,
                                                     const float* __restrict__ beta,
                                                     float* __restrict__ OutF,
                                                     bf16_t* __restrict__ OutB,
                                                     int shift) {
  __shared__ float tile[64][196];
  int w = threadIdx.x >> 5, lane = threadIdx.x & 31;
  int wm = w >> 2, wn = w & 3;                     // 2 M-waves x 4 N-waves (192 = 4*48)
  v8f acc[2][3] = {};
  const bf16_t* Ap0 = A + (long)(blockIdx.x * 64 + wm * 32) * K;
#pragma unroll 3
  for (int kt = 0; kt < K; kt += 32) {
    v16bf a0 = load_afrag(Ap0 + kt, K, lane);
    v16bf a1 = load_afrag(Ap0 + (long)16 * K + kt, K, lane);
#pragma unroll
    for (int f = 0; f < 3; ++f) {
      v16bf b = load_bfrag(BT + (long)(wn * 48 + f * 16) * K + kt, K, lane);
      acc[0][f] = WMMA_BF16(a0, b, acc[0][f]);
      acc[1][f] = WMMA_BF16(a1, b, acc[1][f]);
    }
  }
  int ln = lane & 15, lh = lane >> 4;
#pragma unroll
  for (int mi = 0; mi < 2; ++mi)
#pragma unroll
    for (int f = 0; f < 3; ++f) {
      int col = wn * 48 + f * 16 + ln;
      float bv = bias[col];
#pragma unroll
      for (int j = 0; j < 8; ++j) {
        int row = wm * 32 + mi * 16 + j + lh * 8;
        tile[row][col] = acc[mi][f][j] + bv;
      }
    }
  __syncthreads();
  int t = threadIdx.x;
  if (t < 64) {
    int g;
    if (SCATTER) {
      int wi = blockIdx.x;
      int wis = wi >> 8, wih = (wi >> 4) & 15, wiw = wi & 15;
      int ts = t >> 4, th = (t >> 2) & 3, tw = t & 3;
      int s  = (wis * 4 + ts + shift) & (Sdim - 1);
      int hh = (wih * 4 + th + shift) & (Hdim - 1);
      int ww = (wiw * 4 + tw + shift) & (Wdim - 1);
      g = (s * Hdim + hh) * Wdim + ww;
    } else {
      g = blockIdx.x * 64 + t;
    }
    float mu = 0.f;
    for (int c = 0; c < Cdim; ++c) mu += tile[t][c];
    mu *= (1.f / Cdim);
    float var = 0.f;
    for (int c = 0; c < Cdim; ++c) { float d = tile[t][c] - mu; var += d * d; }
    var *= (1.f / Cdim);
    float inv = rsqrtf(var + 1e-5f);
    const float* sc = shortcut + (long)g * Cdim;
    float* of = OutF + (long)g * Cdim;
    bf16_t* ob = OutB ? OutB + (long)g * Cdim : nullptr;
    for (int c = 0; c < Cdim; ++c) {
      float v = sc[c] + (tile[t][c] - mu) * inv * gamma[c] + beta[c];
      of[c] = v;
      if (ob) ob[c] = f2bf(v);
    }
  }
}

// ---------------- windowed cosine attention, one block per (window, head) ----------------
__global__ __launch_bounds__(128) void k_attn(const bf16_t* __restrict__ qkv,
                                              bf16_t* __restrict__ out,
                                              const float* __restrict__ logit_scale,
                                              const float* __restrict__ rpb,
                                              int layer, int shift) {
  __shared__ float sP[64][64];
  __shared__ __align__(16) bf16_t sVT[32][64];
  __shared__ float sInv[128];   // [0..63] = 1/|q_row|, [64..127] = 1/|k_row|
  __shared__ float sBias[343];  // rel-pos bias table for this head
  __shared__ int sCnt[64];
  int bi = blockIdx.x;
  int wi = bi / HEADS, h = bi % HEADS;
  int t = threadIdx.x;
  const bf16_t* base = qkv + (long)wi * NTOK * 576;  // [64,576]: q|k|v each 192, head stride 32
  {
    int row = t & 63, which = t >> 6;                // which: 0=q, 1=k
    const bf16_t* p = base + (long)row * 576 + which * 192 + h * 32;
    float ss = 0.f;
#pragma unroll
    for (int d = 0; d < 32; ++d) { float v = (float)p[d]; ss += v * v; }
    sInv[t] = 1.f / fmaxf(sqrtf(ss), 1e-12f);
  }
  for (int idx = t; idx < 2048; idx += 128) {        // stage V transposed: sVT[d][token]
    int tok = idx & 63, d = idx >> 6;
    sVT[d][tok] = base[(long)tok * 576 + 384 + h * 32 + d];
  }
  for (int idx = t; idx < 343; idx += 128)           // stage bias table for this head
    sBias[idx] = rpb[(long)(layer * 343 + idx) * HEADS + h];
  if (shift && t < 64) {                             // shifted-window mask region id (rolled frame)
    int wis = wi >> 8, wih = (wi >> 4) & 15, wiw = wi & 15;
    int ts = t >> 4, th = (t >> 2) & 3, tw = t & 3;
    int s = wis * 4 + ts, hh = wih * 4 + th, ww = wiw * 4 + tw;
    int cs = (s  < Sdim - WSZ) ? 0 : ((s  < Sdim - shift) ? 1 : 2);
    int ch = (hh < Hdim - WSZ) ? 0 : ((hh < Hdim - shift) ? 1 : 2);
    int cw = (ww < Wdim - WSZ) ? 0 : ((ww < Wdim - shift) ? 1 : 2);
    sCnt[t] = cs * 9 + ch * 3 + cw;
  }
  __syncthreads();
  int w = t >> 5, lane = t & 31;
  int ln = lane & 15, lh = lane >> 4;
  float scale = __expf(fminf(logit_scale[layer * HEADS + h], 4.6051702f)); // log(100)
  {
    const bf16_t* qp = base + (long)(w * 16) * 576 + h * 32;
    v16bf a = load_afrag(qp, 576, lane);
    const bf16_t* kp = base + 192 + h * 32;
#pragma unroll
    for (int f = 0; f < 4; ++f) {
      v16bf b = load_bfrag(kp + (long)(f * 16) * 576, 576, lane);
      v8f c = {};
      c = WMMA_BF16(a, b, c);
#pragma unroll
      for (int j = 0; j < 8; ++j) {
        int M = w * 16 + j + lh * 8;                 // query token
        int Nn = f * 16 + ln;                        // key token
        float v = c[j] * sInv[M] * sInv[64 + Nn] * scale;
        int dts = (M >> 4) - (Nn >> 4) + 3;          // quirky rel-pos index, matches reference
        int dth = ((M >> 2) & 3) - ((Nn >> 2) & 3) + 3;
        int dtw = (M & 3) - (Nn & 3) + 3;
        v += sBias[dts * 11 + dth * 7 + dtw];
        if (shift && (sCnt[M] != sCnt[Nn])) v -= 100.f;
        sP[M][Nn] = v;
      }
    }
  }
  __syncthreads();
  if (t < 64) {                                      // softmax over keys
    float mx = -1e30f;
    for (int j = 0; j < 64; ++j) mx = fmaxf(mx, sP[t][j]);
    float sum = 0.f;
    for (int j = 0; j < 64; ++j) { float e = __expf(sP[t][j] - mx); sP[t][j] = e; sum += e; }
    float r = 1.f / sum;
    for (int j = 0; j < 64; ++j) sP[t][j] *= r;
  }
  __syncthreads();
  v8f o0 = {}, o1 = {};
#pragma unroll
  for (int kt = 0; kt < 2; ++kt) {                   // P[64x64] @ V[64x32]
    v16bf a  = load_afrag_f32(&sP[w * 16][0], 64, lane, kt * 32);
    v16bf b0 = load_bfrag(&sVT[0][0] + kt * 32, 64, lane);
    v16bf b1 = load_bfrag(&sVT[16][0] + kt * 32, 64, lane);
    o0 = WMMA_BF16(a, b0, o0);
    o1 = WMMA_BF16(a, b1, o1);
  }
#pragma unroll
  for (int f = 0; f < 2; ++f) {
    v8f c = f ? o1 : o0;
#pragma unroll
    for (int j = 0; j < 8; ++j) {
      int M = w * 16 + j + lh * 8;
      int d = f * 16 + ln;
      out[((long)wi * NTOK + M) * Cdim + h * 32 + d] = f2bf(c[j]);
    }
  }
}

// ---------------- launcher ----------------
extern "C" void kernel_launch(void* const* d_in, const int* in_sizes, int n_in,
                              void* d_out, int out_size, void* d_ws, size_t ws_size,
                              hipStream_t stream) {
  const float* x      = (const float*)d_in[0];
  const float* qkv_w  = (const float*)d_in[1];
  const float* qkv_b  = (const float*)d_in[2];
  const float* proj_w = (const float*)d_in[3];
  const float* proj_b = (const float*)d_in[4];
  const float* lsc    = (const float*)d_in[5];
  const float* rpb    = (const float*)d_in[6];
  const float* n1_g   = (const float*)d_in[7];
  const float* n1_b   = (const float*)d_in[8];
  const float* n2_g   = (const float*)d_in[9];
  const float* n2_b   = (const float*)d_in[10];
  const float* fc1_w  = (const float*)d_in[11];
  const float* fc1_b  = (const float*)d_in[12];
  const float* fc2_w  = (const float*)d_in[13];
  const float* fc2_b  = (const float*)d_in[14];
  float* outx = (float*)d_out;

  char* ws = (char*)d_ws;
  size_t off = 0;
  bf16_t* wT   = (bf16_t*)(ws + off); off += (size_t)1 << 20;
  bf16_t* bufA = (bf16_t*)(ws + off); off += (size_t)NWIN * NTOK * Cdim * 2;  // 48 MB: xw / attn_out
  bf16_t* bufB = (bf16_t*)(ws + off); off += (size_t)MTOT * HID * 2;          // 192 MB: qkv / h1
  float*  xA   = (float*) (ws + off); off += (size_t)MTOT * Cdim * 4;         // post-attn x (f32)
  bf16_t* xAb  = (bf16_t*)(ws + off); off += (size_t)MTOT * Cdim * 2;         // post-attn x (bf16)
  float*  xbuf = (float*) (ws + off); off += (size_t)MTOT * Cdim * 4;         // layer-0 output x

  bf16_t* qkvT  = wT;
  bf16_t* projT = qkvT + 576 * 192;
  bf16_t* fc1T  = projT + 192 * 192;
  bf16_t* fc2T  = fc1T + 768 * 192;
  bf16_t* xw      = bufA;   // dead after QKV GEMM
  bf16_t* attno   = bufA;   // written by attention (aliases xw)
  bf16_t* qkvbuf  = bufB;   // dead after attention
  bf16_t* h1      = bufB;   // written by FC1 (aliases qkv)

  for (int layer = 0; layer < 2; ++layer) {
    int shift = (layer & 1) ? (WSZ / 2) : 0;
    const float* xin = (layer == 0) ? x : xbuf;
    float* xout = (layer == 1) ? outx : xbuf;

    k_transpose<<<(192 * 576 + 255) / 256, 256, 0, stream>>>(qkv_w + (size_t)layer * 192 * 576, qkvT, 192, 576);
    k_transpose<<<(192 * 192 + 255) / 256, 256, 0, stream>>>(proj_w + (size_t)layer * 192 * 192, projT, 192, 192);
    k_transpose<<<(192 * 768 + 255) / 256, 256, 0, stream>>>(fc1_w + (size_t)layer * 192 * 768, fc1T, 192, 768);
    k_transpose<<<(768 * 192 + 255) / 256, 256, 0, stream>>>(fc2_w + (size_t)layer * 768 * 192, fc2T, 768, 192);

    k_gather<<<(NWIN * NTOK * Cdim) / 256, 256, 0, stream>>>(xin, xw, shift);

    dim3 gq(MTOT / 128, 576 / 64);
    k_gemm_tile<0, 192><<<gq, 256, 0, stream>>>(xw, qkvT, qkv_b + (size_t)layer * 576, qkvbuf, 576);

    k_attn<<<NWIN * HEADS, 128, 0, stream>>>(qkvbuf, attno, lsc, rpb, layer, shift);

    k_gemm_row_ln<1, 192><<<NWIN, 256, 0, stream>>>(attno, projT, proj_b + (size_t)layer * 192,
        xin, n1_g + (size_t)layer * 192, n1_b + (size_t)layer * 192, xA, xAb, shift);

    dim3 gf1(MTOT / 128, HID / 64);
    k_gemm_tile<1, 192><<<gf1, 256, 0, stream>>>(xAb, fc1T, fc1_b + (size_t)layer * HID, h1, HID);

    k_gemm_row_ln<0, 768><<<MTOT / 64, 256, 0, stream>>>(h1, fc2T, fc2_b + (size_t)layer * 192,
        xA, n2_g + (size_t)layer * 192, n2_b + (size_t)layer * 192, xout, nullptr, 0);
  }
}
